// PointASISTransformer_All_attribute_30193620091452
// MI455X (gfx1250) — matrix operand order repaired and checked by reading, not compiled, not run on hardware
//
#include <hip/hip_runtime.h>
#include <hip/hip_bf16.h>
#include <math.h>

#define BB 8
#define NPTS 16384
#define NATTR 14
#define NG 256
#define NM 32
#define NENC 1024
#define NDIM 384
#define NDEPTH 12
#define NHEADS 6
#define NHID 1536
#define QKVLD (3 * NDIM)
#define NROWS (BB * NG * NM)   // 65536 points through encoder
#define NBG (BB * NG)          // 2048 groups / tokens
#define HD (NDIM / NHEADS)     // 64
#define AS_COLS 40             // 32 + pad (avoids LDS bank conflicts)

typedef __attribute__((ext_vector_type(16))) _Float16 v16h;
typedef __attribute__((ext_vector_type(8)))  float    v8f;

// ---------------- WMMA fragment helpers (wave32, v_wmma_f32_16x16x32_f16) ---
// A fragment from global: 16x32 (MxK). lane L: m=L&15, half=L>>4.
// v16h elems [0..7]  = A[m][kbase + 16*half + 0..7]
// v16h elems [8..15] = A[m][kbase + 16*half + 8 + 0..7]
template<typename T>
__device__ inline v16h frag_a(const T* A, int lda, int kbase, int K) {
  int lane = threadIdx.x & 31;
  int m = lane & 15, hl = lane >> 4;
  const T* row = A + (size_t)m * lda;
  v16h f;
#pragma unroll
  for (int i = 0; i < 8; ++i) {
    int k0 = kbase + 16 * hl + i;
    int k1 = kbase + 16 * hl + 8 + i;
    f[i]     = (k0 < K) ? (_Float16)row[k0] : (_Float16)0.f;
    f[i + 8] = (k1 < K) ? (_Float16)row[k1] : (_Float16)0.f;
  }
  return f;
}

// A fragment from an LDS-staged 16x32 tile (already zero-padded on K edge).
__device__ inline v16h frag_a_lds(const _Float16 As[16][AS_COLS]) {
  int lane = threadIdx.x & 31;
  int m = lane & 15, hl = lane >> 4;
  v16h f;
#pragma unroll
  for (int i = 0; i < 8; ++i) {
    f[i]     = As[m][16 * hl + i];
    f[i + 8] = As[m][16 * hl + 8 + i];
  }
  return f;
}

// B fragment: 32x16 (KxN), B[k][n] = W[n*ldw + k], W already f16 (K-contiguous rows).
// lane L: n=L&15, half=L>>4; v16h elems [0..15] = B[kbase + 16*half + 0..15][n]
__device__ inline v16h frag_b16(const _Float16* W, int ldw, int kbase, int K, int nvalid) {
  int lane = threadIdx.x & 31;
  int n = lane & 15, hl = lane >> 4;
  v16h f;
  if (n >= nvalid) {
#pragma unroll
    for (int i = 0; i < 16; ++i) f[i] = (_Float16)0.f;
    return f;
  }
  const _Float16* row = W + (size_t)n * ldw;
  if (kbase + 32 <= K) {
    const _Float16* p = row + kbase + 16 * hl;
#pragma unroll
    for (int i = 0; i < 16; ++i) f[i] = p[i];
  } else {
#pragma unroll
    for (int i = 0; i < 16; ++i) {
      int k = kbase + 16 * hl + i;
      f[i] = (k < K) ? row[k] : (_Float16)0.f;
    }
  }
  return f;
}

// B fragment with B[k][n] = Bm[k*ldb + n] (column n strided by ldb).
template<typename T>
__device__ inline v16h frag_b_col(const T* Bm, int ldb, int kbase, int K, int nvalid) {
  int lane = threadIdx.x & 31;
  int n = lane & 15, hl = lane >> 4;
  v16h f;
  if (n >= nvalid) {
#pragma unroll
    for (int i = 0; i < 16; ++i) f[i] = (_Float16)0.f;
    return f;
  }
#pragma unroll
  for (int i = 0; i < 16; ++i) {
    int k = kbase + 16 * hl + i;
    f[i] = (k < K) ? (_Float16)Bm[(size_t)k * ldb + n] : (_Float16)0.f;
  }
  return f;
}

__device__ inline float activate(float v, int act) {
  if (act == 1) return v > 0.f ? v : 0.f;
  if (act == 2) {  // tanh-approx gelu (jax.nn.gelu default)
    float u = 0.7978845608028654f * (v + 0.044715f * v * v * v);
    return 0.5f * v * (1.f + tanhf(u));
  }
  return v;
}

// C/D fragment layout: lane L: n=L&15, half=L>>4; elem r -> C[r + 8*half][n]
template<typename OT>
__device__ inline void store_tile(OT* Y, int ldy, v8f acc, int nvalid,
                                  const float* bias, const float* res, int ldres,
                                  int act, float alpha) {
  int lane = threadIdx.x & 31;
  int n = lane & 15, hl = lane >> 4;
  if (n >= nvalid) return;
  float bb = bias ? bias[n] : 0.f;
#pragma unroll
  for (int r = 0; r < 8; ++r) {
    int m = r + 8 * hl;
    float v = acc[r] * alpha + bb;
    v = activate(v, act);
    if (res) v += res[(size_t)m * ldres + n];
    Y[(size_t)m * ldy + n] = (OT)v;
  }
}

// Generic GEMM: Y[m][n] = act(alpha * sum_k A[m][k]*W[n][k] + bias[n]) + res[m][n]
// W is pre-converted f16. Block = 8 waves sharing one LDS-staged 16-row A tile;
// each wave owns a 16x64 strip. grid.x = M/16, grid.y = ceil(N/512).
// NOTE: no wave early-returns (all must reach __syncthreads).
template<typename AT, typename OT>
__global__ void gemm_wrow(const AT* A, int lda,
                          const _Float16* W, int ldw,
                          const float* bias,
                          const float* res, int ldres,
                          OT* Y, int ldy,
                          int M, int N, int K, int act, float alpha) {
  __shared__ _Float16 As[16][AS_COLS];
  int m0 = blockIdx.x << 4;
  int wave = threadIdx.x >> 5;
  int n0 = (blockIdx.y * 8 + wave) << 6;
  bool active = (n0 < N);
  bool full = active && (N - n0 >= 64);
  v8f acc[4];
#pragma unroll
  for (int t = 0; t < 4; ++t)
#pragma unroll
    for (int r = 0; r < 8; ++r) acc[t][r] = 0.f;

  int sr = threadIdx.x >> 4;         // 0..15 : staged row
  int sc = (threadIdx.x & 15) << 1;  // 0,2,..,30 : staged col pair
  const AT* arow = A + (size_t)(m0 + sr) * lda;

  for (int k = 0; k < K; k += 32) {
    __syncthreads();
    {
      int k0 = k + sc;
      float v0 = (k0 < K) ? (float)arow[k0] : 0.f;
      float v1 = (k0 + 1 < K) ? (float)arow[k0 + 1] : 0.f;
      As[sr][sc] = (_Float16)v0;
      As[sr][sc + 1] = (_Float16)v1;
    }
    __syncthreads();
    if (!active) continue;
    v16h a = frag_a_lds(As);
    if (full) {
#pragma unroll
      for (int t = 0; t < 4; ++t) {
        v16h b = frag_b16(W + (size_t)(n0 + (t << 4)) * ldw, ldw, k, K, 16);
        acc[t] = __builtin_amdgcn_wmma_f32_16x16x32_f16(false, a, false, b,
                                                        (short)0, acc[t], false, false);
      }
    } else {
#pragma unroll
      for (int t = 0; t < 4; ++t) {
        int nt = n0 + (t << 4);
        if (nt < N) {
          v16h b = frag_b16(W + (size_t)nt * ldw, ldw, k, K, N - nt);
          acc[t] = __builtin_amdgcn_wmma_f32_16x16x32_f16(false, a, false, b,
                                                          (short)0, acc[t], false, false);
        }
      }
    }
  }
  if (!active) return;
#pragma unroll
  for (int t = 0; t < 4; ++t) {
    int nt = n0 + (t << 4);
    if (nt < N)
      store_tile<OT>(Y + (size_t)m0 * ldy + nt, ldy, acc[t], N - nt,
                     bias ? bias + nt : nullptr,
                     res ? res + (size_t)m0 * ldres + nt : nullptr, ldres,
                     act, alpha);
  }
}

// ------------------------- attention (WMMA) ---------------------------------
// scores[z][m][n] = scale * sum_d q[m][d]*k[n][d]; z = b*NHEADS + h; qkv is f16
__global__ void attn_scores_ker(const _Float16* qkv, float* S, float scale) {
  int z = blockIdx.z;
  int b = z / NHEADS, h = z % NHEADS;
  int m0 = blockIdx.x << 4;
  int wave = threadIdx.x >> 5;
  int n0 = wave << 6;  // 4 waves cover 256 cols
  const _Float16* Q  = qkv + (size_t)b * NG * QKVLD + h * HD;
  const _Float16* Kp = qkv + (size_t)b * NG * QKVLD + NDIM + h * HD;
  const _Float16* Ab = Q + (size_t)m0 * QKVLD;
  v8f acc[4];
#pragma unroll
  for (int t = 0; t < 4; ++t)
#pragma unroll
    for (int r = 0; r < 8; ++r) acc[t][r] = 0.f;
  for (int k = 0; k < HD; k += 32) {
    v16h a = frag_a<_Float16>(Ab, QKVLD, k, HD);
#pragma unroll
    for (int t = 0; t < 4; ++t) {
      int nt = n0 + (t << 4);
      v16h bfr = frag_b16(Kp + (size_t)nt * QKVLD, QKVLD, k, HD, 16);
      acc[t] = __builtin_amdgcn_wmma_f32_16x16x32_f16(false, a, false, bfr,
                                                      (short)0, acc[t], false, false);
    }
  }
  float* Yb = S + (size_t)z * NG * NG + (size_t)m0 * NG;
#pragma unroll
  for (int t = 0; t < 4; ++t) {
    int nt = n0 + (t << 4);
    store_tile<float>(Yb + nt, NG, acc[t], 16, nullptr, nullptr, 0, 0, scale);
  }
}

// out[token m][h*64+d] = sum_n a[m][n] * v[n][d]; output f16
__global__ void attn_out_ker(const float* S, const _Float16* qkv, _Float16* attno) {
  int z = blockIdx.z;
  int b = z / NHEADS, h = z % NHEADS;
  int m0 = blockIdx.x << 4;
  const float* A = S + (size_t)z * NG * NG + (size_t)m0 * NG;
  const _Float16* V = qkv + (size_t)b * NG * QKVLD + 2 * NDIM + h * HD;
  v8f acc[4];
#pragma unroll
  for (int t = 0; t < 4; ++t)
#pragma unroll
    for (int r = 0; r < 8; ++r) acc[t][r] = 0.f;
  for (int k = 0; k < NG; k += 32) {
    v16h a = frag_a<float>(A, NG, k, NG);
#pragma unroll
    for (int t = 0; t < 4; ++t) {
      v16h bfr = frag_b_col<_Float16>(V + (t << 4), QKVLD, k, NG, 16);
      acc[t] = __builtin_amdgcn_wmma_f32_16x16x32_f16(false, a, false, bfr,
                                                      (short)0, acc[t], false, false);
    }
  }
  _Float16* Y = attno + ((size_t)b * NG + m0) * NDIM + h * HD;
#pragma unroll
  for (int t = 0; t < 4; ++t)
    store_tile<_Float16>(Y + (t << 4), NDIM, acc[t], 16, nullptr, nullptr, 0, 0, 1.f);
}

// ------------------------- scalar kernels -----------------------------------
__device__ inline float block_reduce_sum(float v, float* sbuf) {
  int t = threadIdx.x;
  sbuf[t] = v; __syncthreads();
  for (int s = blockDim.x >> 1; s > 0; s >>= 1) {
    if (t < s) sbuf[t] += sbuf[t + s];
    __syncthreads();
  }
  float r = sbuf[0];
  __syncthreads();
  return r;
}

__global__ void cvt_f16_ker(const float* src, _Float16* dst, int n) {
  int i = blockIdx.x * blockDim.x + threadIdx.x;
  if (i < n) dst[i] = (_Float16)src[i];
}

// Farthest point sampling: one block per batch, 256 iterative argmax steps.
__global__ void fps_ker(const float* xyz, float* dist, int* cidx) {
  int b = blockIdx.x;
  const float* X = xyz + (size_t)b * NPTS * NATTR;
  float* D = dist + (size_t)b * NPTS;
  __shared__ float bv[256];
  __shared__ int   bi[256];
  __shared__ int   s_far;
  if (threadIdx.x == 0) s_far = 0;
  __syncthreads();
  for (int it = 0; it < NG; ++it) {
    int far = s_far;
    if (threadIdx.x == 0) cidx[b * NG + it] = far;
    float cx = X[(size_t)far * NATTR + 0];
    float cy = X[(size_t)far * NATTR + 1];
    float cz = X[(size_t)far * NATTR + 2];
    float best = -1.f; int bidx = 0;
    for (int p = threadIdx.x; p < NPTS; p += 256) {
      float dx = X[(size_t)p * NATTR + 0] - cx;
      float dy = X[(size_t)p * NATTR + 1] - cy;
      float dz = X[(size_t)p * NATTR + 2] - cz;
      float d = dx * dx + dy * dy + dz * dz;
      float old = (it == 0) ? 1e10f : D[p];
      float nd = fminf(old, d);
      D[p] = nd;
      if (nd > best) { best = nd; bidx = p; }
    }
    bv[threadIdx.x] = best; bi[threadIdx.x] = bidx;
    __syncthreads();
    for (int s = 128; s > 0; s >>= 1) {
      if (threadIdx.x < s) {
        float ov = bv[threadIdx.x + s]; int oi = bi[threadIdx.x + s];
        if (ov > bv[threadIdx.x] || (ov == bv[threadIdx.x] && oi < bi[threadIdx.x])) {
          bv[threadIdx.x] = ov; bi[threadIdx.x] = oi;
        }
      }
      __syncthreads();
    }
    if (threadIdx.x == 0) s_far = bi[0];
    __syncthreads();
  }
}

__global__ void gather_centers_ker(const float* xyz, const int* cidx,
                                   float* cattrs, float* center) {
  int bg = blockIdx.x;
  int b = bg >> 8;
  int idx = cidx[bg];
  const float* src = xyz + ((size_t)b * NPTS + idx) * NATTR;
  if (threadIdx.x < NATTR) cattrs[(size_t)bg * NATTR + threadIdx.x] = src[threadIdx.x];
  if (threadIdx.x < 3) center[bg * 3 + threadIdx.x] = src[threadIdx.x];
}

// kNN via 32 iterative argmin passes over a per-group d2 scratch row.
__global__ void knn_ker(const float* xyz, const float* cattrs, float* d2buf, int* knn) {
  int bg = blockIdx.x;
  int b = bg >> 8;
  const float* X = xyz + (size_t)b * NPTS * NATTR;
  float* D = d2buf + (size_t)bg * NPTS;
  __shared__ float c[NATTR];
  __shared__ float bv[256];
  __shared__ int   bi[256];
  if (threadIdx.x < NATTR) c[threadIdx.x] = cattrs[bg * NATTR + threadIdx.x];
  __syncthreads();
  for (int p = threadIdx.x; p < NPTS; p += 256) {
    float s = 0.f;
    for (int a = 0; a < NATTR; ++a) {
      float d = X[(size_t)p * NATTR + a] - c[a];
      s += d * d;
    }
    D[p] = s;
  }
  __syncthreads();
  for (int it = 0; it < NM; ++it) {
    float best = 3.9e38f; int bidx = 0;
    for (int p = threadIdx.x; p < NPTS; p += 256) {
      float v = D[p];
      if (v < best) { best = v; bidx = p; }
    }
    bv[threadIdx.x] = best; bi[threadIdx.x] = bidx;
    __syncthreads();
    for (int s = 128; s > 0; s >>= 1) {
      if (threadIdx.x < s) {
        float ov = bv[threadIdx.x + s]; int oi = bi[threadIdx.x + s];
        if (ov < bv[threadIdx.x] || (ov == bv[threadIdx.x] && oi < bi[threadIdx.x])) {
          bv[threadIdx.x] = ov; bi[threadIdx.x] = oi;
        }
      }
      __syncthreads();
    }
    if (threadIdx.x == 0) { knn[bg * NM + it] = bi[0]; D[bi[0]] = 3.9e38f; }
    __syncthreads();
  }
}

__global__ void gather_nb_ker(const float* xyz, const int* knn, const float* center,
                              float* pg) {
  int bg = blockIdx.x;
  int b = bg >> 8;
  for (int e = threadIdx.x; e < NM * NATTR; e += blockDim.x) {
    int m = e / NATTR, ccol = e % NATTR;
    int idx = knn[bg * NM + m];
    float v = xyz[((size_t)b * NPTS + idx) * NATTR + ccol];
    if (ccol < 3) v -= center[bg * 3 + ccol];
    pg[(size_t)bg * NM * NATTR + e] = v;
  }
}

template<typename T>
__global__ void bn_stats_ker(const T* X, int ld, int rows, float* mean, float* var) {
  int c = blockIdx.x;
  __shared__ float sbuf[256];
  float s = 0.f, ss = 0.f;
  for (int r = threadIdx.x; r < rows; r += 256) {
    float v = (float)X[(size_t)r * ld + c];
    s += v; ss += v * v;
  }
  float ts = block_reduce_sum(s, sbuf);
  float tss = block_reduce_sum(ss, sbuf);
  if (threadIdx.x == 0) {
    float m = ts / rows;
    mean[c] = m;
    var[c] = tss / rows - m * m;
  }
}

template<typename T>
__global__ void bn_apply_ker(T* X, int C, size_t total, const float* mean, const float* var,
                             const float* g, const float* b, int relu) {
  size_t i = (size_t)blockIdx.x * blockDim.x + threadIdx.x;
  size_t stride = (size_t)gridDim.x * blockDim.x;
  for (; i < total; i += stride) {
    int c = (int)(i % C);
    float v = (float)X[i];
    v = (v - mean[c]) * rsqrtf(var[c] + 1e-5f) * g[c] + b[c];
    if (relu) v = fmaxf(v, 0.f);
    X[i] = (T)v;
  }
}

template<typename T>
__global__ void group_max_ker(const T* X, int ld, int Mg, float* Y, int C) {
  int gidx = blockIdx.x;
  const T* base = X + (size_t)gidx * Mg * ld;
  for (int c = threadIdx.x; c < C; c += blockDim.x) {
    float mx = -3.4e38f;
    for (int m = 0; m < Mg; ++m) mx = fmaxf(mx, (float)base[(size_t)m * ld + c]);
    Y[(size_t)gidx * C + c] = mx;
  }
}

__global__ void add_eye_ker(float* tmat, int n) {
  int i = blockIdx.x * blockDim.x + threadIdx.x;
  if (i < n) {
    int r = i % 36;
    if (r % 7 == 0) tmat[i] += 1.f;  // positions 0,7,14,21,28,35 = diagonal
  }
}

// enc_in[:, :6] = x6 @ t (per group); enc_in[:, 6:] = pg[:, 6:]
__global__ void transform_ker(const float* pg, const float* tmat, _Float16* enc_in) {
  int bg = blockIdx.x;
  const float* P = pg + (size_t)bg * NM * NATTR;
  __shared__ float ts[36];
  if (threadIdx.x < 36) ts[threadIdx.x] = tmat[bg * 36 + threadIdx.x];
  __syncthreads();
  for (int e = threadIdx.x; e < NM * NATTR; e += blockDim.x) {
    int m = e / NATTR, c = e % NATTR;
    float v;
    if (c < 6) {
      v = 0.f;
      for (int j = 0; j < 6; ++j) v += P[m * NATTR + j] * ts[j * 6 + c];
    } else {
      v = P[e];
    }
    enc_in[(size_t)bg * NM * NATTR + e] = (_Float16)v;
  }
}

// LayerNorm: f32 in, f16 out (feeds WMMA A operand)
__global__ void layernorm_ker(const float* X, const float* g, const float* b,
                              _Float16* Y, int C) {
  int row = blockIdx.x;
  const float* x = X + (size_t)row * C;
  _Float16* y = Y + (size_t)row * C;
  __shared__ float sbuf[256];
  float s = 0.f, ss = 0.f;
  for (int c = threadIdx.x; c < C; c += 256) {
    float v = x[c]; s += v; ss += v * v;
  }
  float ts = block_reduce_sum(s, sbuf);
  float tss = block_reduce_sum(ss, sbuf);
  float m = ts / C;
  float vv = tss / C - m * m;
  float inv = rsqrtf(vv + 1e-5f);
  for (int c = threadIdx.x; c < C; c += 256)
    y[c] = (_Float16)((x[c] - m) * inv * g[c] + b[c]);
}

__global__ void softmax_ker(float* S, int cols) {
  int row = blockIdx.x;
  float* x = S + (size_t)row * cols;
  __shared__ float sbuf[256];
  float mx = -3.4e38f;
  for (int c = threadIdx.x; c < cols; c += 256) mx = fmaxf(mx, x[c]);
  sbuf[threadIdx.x] = mx; __syncthreads();
  for (int s = 128; s > 0; s >>= 1) {
    if (threadIdx.x < s) sbuf[threadIdx.x] = fmaxf(sbuf[threadIdx.x], sbuf[threadIdx.x + s]);
    __syncthreads();
  }
  float rowmax = sbuf[0]; __syncthreads();
  float s = 0.f;
  for (int c = threadIdx.x; c < cols; c += 256) {
    float e = expf(x[c] - rowmax);
    x[c] = e; s += e;
  }
  float tot = block_reduce_sum(s, sbuf);
  float inv = 1.f / tot;
  for (int c = threadIdx.x; c < cols; c += 256) x[c] *= inv;
}

__global__ void add_ker(const float* a, const float* b, float* y, int n) {
  int i = blockIdx.x * blockDim.x + threadIdx.x;
  if (i < n) y[i] = a[i] + b[i];
}

// ----------------------------- host orchestration ---------------------------
extern "C" void kernel_launch(void* const* d_in, const int* in_sizes, int n_in,
                              void* d_out, int out_size, void* d_ws, size_t ws_size,
                              hipStream_t stream) {
  const float* xyz      = (const float*)d_in[0];
  const float* tn_c1_w  = (const float*)d_in[1];
  const float* tn_c1_b  = (const float*)d_in[2];
  const float* tn_bn1_g = (const float*)d_in[3];
  const float* tn_bn1_b = (const float*)d_in[4];
  const float* tn_c2_w  = (const float*)d_in[5];
  const float* tn_c2_b  = (const float*)d_in[6];
  const float* tn_bn2_g = (const float*)d_in[7];
  const float* tn_bn2_b = (const float*)d_in[8];
  const float* tn_fc_w  = (const float*)d_in[9];
  const float* tn_fc_b  = (const float*)d_in[10];
  const float* e_w[4]  = {(const float*)d_in[11], (const float*)d_in[15],
                          (const float*)d_in[19], (const float*)d_in[23]};
  const float* e_b[4]  = {(const float*)d_in[12], (const float*)d_in[16],
                          (const float*)d_in[20], (const float*)d_in[24]};
  const float* e_g[4]  = {(const float*)d_in[13], (const float*)d_in[17],
                          (const float*)d_in[21], (const float*)d_in[25]};
  const float* e_bb[4] = {(const float*)d_in[14], (const float*)d_in[18],
                          (const float*)d_in[22], (const float*)d_in[26]};
  const float* reduce_w = (const float*)d_in[27];
  const float* reduce_b = (const float*)d_in[28];
  const float* pos1_w   = (const float*)d_in[29];
  const float* pos1_b   = (const float*)d_in[30];
  const float* pos2_w   = (const float*)d_in[31];
  const float* pos2_b   = (const float*)d_in[32];
  const float* ln1_g    = (const float*)d_in[33];
  const float* ln1_b    = (const float*)d_in[34];
  const float* qkv_w    = (const float*)d_in[35];
  const float* proj_w   = (const float*)d_in[36];
  const float* proj_b   = (const float*)d_in[37];
  const float* ln2_g    = (const float*)d_in[38];
  const float* ln2_b    = (const float*)d_in[39];
  const float* fc1_w    = (const float*)d_in[40];
  const float* fc1_b    = (const float*)d_in[41];
  const float* fc2_w    = (const float*)d_in[42];
  const float* fc2_b    = (const float*)d_in[43];
  (void)n_in; (void)in_sizes; (void)out_size;

  // ---- workspace layout (time-aliased) ----
  char* w = (char*)d_ws;
  size_t used = 0;
  auto alloc = [&](size_t bytes) -> char* {
    char* p = w;
    size_t pad = (bytes + 255) & ~(size_t)255;
    w += pad; used += pad;
    return p;
  };
  float*     fps_dist = (float*)alloc((size_t)BB * NPTS * 4);
  int*       cidx     = (int*)  alloc((size_t)BB * NG * 4);
  float*     cattrs   = (float*)alloc((size_t)NBG * NATTR * 4);
  float*     center   = (float*)alloc((size_t)NBG * 3 * 4);
  int*       knn      = (int*)  alloc((size_t)NBG * NM * 4);
  float*     pg       = (float*)alloc((size_t)NBG * NM * NATTR * 4);
  _Float16*  enc_in   = (_Float16*)alloc((size_t)NROWS * NATTR * 2);
  // BIG_A: kNN d2 scratch (2048*16384 f32) aliases conv4 output (65536*1024 f16) = 128MB each
  char*      bigA     = alloc((size_t)NROWS * NENC * 2);
  _Float16*  bigB     = (_Float16*)alloc((size_t)NROWS * 512 * 2); // tnet h2 / p3
  _Float16*  bigC     = (_Float16*)alloc((size_t)NROWS * 256 * 2); // p2
  _Float16*  bigD     = (_Float16*)alloc((size_t)NROWS * 128 * 2); // tnet h1 / p1
  float*     hmax     = (float*)alloc((size_t)NBG * 512 * 4);
  float*     tmat     = (float*)alloc((size_t)NBG * 36 * 4);
  float*     tokens   = (float*)alloc((size_t)NBG * NENC * 4);
  float*     xbuf     = (float*)alloc((size_t)NBG * NDIM * 4);
  float*     posbuf   = (float*)alloc((size_t)NBG * NDIM * 4);
  _Float16*  poshid   = (_Float16*)alloc((size_t)NBG * 128 * 2);
  float*     hin      = (float*)alloc((size_t)NBG * NDIM * 4);
  _Float16*  ybuf     = (_Float16*)alloc((size_t)NBG * NDIM * 2);
  _Float16*  qkvbuf   = (_Float16*)alloc((size_t)NBG * QKVLD * 2);
  float*     scores   = (float*)alloc((size_t)BB * NHEADS * NG * NG * 4);
  _Float16*  attno    = (_Float16*)alloc((size_t)NBG * NDIM * 2);
  _Float16*  mlp      = (_Float16*)alloc((size_t)NBG * NHID * 2);
  float*     bnmean   = (float*)alloc(1024 * 4);
  float*     bnvar    = (float*)alloc(1024 * 4);

  // f16 weight staging (converted once per launch)
  struct WCvt { const float* src; _Float16* dst; size_t n; };
  WCvt wc[16]; int nwc = 0;
  auto cvtw = [&](const float* src, size_t n) -> _Float16* {
    _Float16* dst = (_Float16*)alloc(n * 2);
    wc[nwc].src = src; wc[nwc].dst = dst; wc[nwc].n = n; ++nwc;
    return dst;
  };
  _Float16* tn_c1_w16 = cvtw(tn_c1_w, (size_t)64 * 6);
  _Float16* tn_c2_w16 = cvtw(tn_c2_w, (size_t)512 * 64);
  _Float16* tn_fc_w16 = cvtw(tn_fc_w, (size_t)36 * 512);
  _Float16* e_w16[4];
  e_w16[0] = cvtw(e_w[0], (size_t)128 * NATTR);
  e_w16[1] = cvtw(e_w[1], (size_t)256 * 128);
  e_w16[2] = cvtw(e_w[2], (size_t)512 * 256);
  e_w16[3] = cvtw(e_w[3], (size_t)NENC * 512);
  _Float16* reduce_w16 = cvtw(reduce_w, (size_t)NDIM * NENC);
  _Float16* pos1_w16   = cvtw(pos1_w, (size_t)128 * 3);
  _Float16* pos2_w16   = cvtw(pos2_w, (size_t)NDIM * 128);
  _Float16* qkv_w16    = cvtw(qkv_w, (size_t)NDEPTH * QKVLD * NDIM);
  _Float16* proj_w16   = cvtw(proj_w, (size_t)NDEPTH * NDIM * NDIM);
  _Float16* fc1_w16    = cvtw(fc1_w, (size_t)NDEPTH * NHID * NDIM);
  _Float16* fc2_w16    = cvtw(fc2_w, (size_t)NDEPTH * NDIM * NHID);
  if (used > ws_size) return;  // workspace too small: bail out safely

  for (int i = 0; i < nwc; ++i)
    cvt_f16_ker<<<(int)((wc[i].n + 255) / 256), 256, 0, stream>>>(
        wc[i].src, wc[i].dst, (int)wc[i].n);

  float*    d2buf = (float*)bigA;     // kNN phase
  _Float16* p4    = (_Float16*)bigA;  // encoder phase (d2buf dead by then)

  const int GEMM_T = 256;                 // 8 waves per block
  dim3 gRows(NROWS / 16, 1), gTok(NBG / 16, 1);

  // ---- stage 1: sampling / grouping ----
  fps_ker<<<BB, 256, 0, stream>>>(xyz, fps_dist, cidx);
  gather_centers_ker<<<NBG, 32, 0, stream>>>(xyz, cidx, cattrs, center);
  knn_ker<<<NBG, 256, 0, stream>>>(xyz, cattrs, d2buf, knn);
  gather_nb_ker<<<NBG, 256, 0, stream>>>(xyz, knn, center, pg);

  // ---- stage 2: TNet ----
  _Float16* h1 = bigD;  // 65536 x 64
  gemm_wrow<float, _Float16><<<gRows, GEMM_T, 0, stream>>>(
      pg, NATTR, tn_c1_w16, 6, tn_c1_b, nullptr, 0, h1, 64, NROWS, 64, 6, 0, 1.f);
  bn_stats_ker<_Float16><<<64, 256, 0, stream>>>(h1, 64, NROWS, bnmean, bnvar);
  bn_apply_ker<_Float16><<<4096, 256, 0, stream>>>(h1, 64, (size_t)NROWS * 64,
                                                   bnmean, bnvar, tn_bn1_g, tn_bn1_b, 1);
  _Float16* h2 = bigB;  // 65536 x 512
  gemm_wrow<_Float16, _Float16><<<gRows, GEMM_T, 0, stream>>>(
      h1, 64, tn_c2_w16, 64, tn_c2_b, nullptr, 0, h2, 512, NROWS, 512, 64, 0, 1.f);
  bn_stats_ker<_Float16><<<512, 256, 0, stream>>>(h2, 512, NROWS, bnmean, bnvar);
  bn_apply_ker<_Float16><<<8192, 256, 0, stream>>>(h2, 512, (size_t)NROWS * 512,
                                                   bnmean, bnvar, tn_bn2_g, tn_bn2_b, 1);
  group_max_ker<_Float16><<<NBG, 256, 0, stream>>>(h2, 512, NM, hmax, 512);
  gemm_wrow<float, float><<<gTok, GEMM_T, 0, stream>>>(
      hmax, 512, tn_fc_w16, 512, tn_fc_b, nullptr, 0, tmat, 36, NBG, 36, 512, 0, 1.f);
  add_eye_ker<<<(NBG * 36 + 255) / 256, 256, 0, stream>>>(tmat, NBG * 36);
  transform_ker<<<NBG, 256, 0, stream>>>(pg, tmat, enc_in);

  // ---- stage 3: encoder convs with batch-stat BN ----
  _Float16* p1 = bigD;  // 65536 x 128
  gemm_wrow<_Float16, _Float16><<<gRows, GEMM_T, 0, stream>>>(
      enc_in, NATTR, e_w16[0], NATTR, e_b[0], nullptr, 0, p1, 128, NROWS, 128, NATTR, 0, 1.f);
  bn_stats_ker<_Float16><<<128, 256, 0, stream>>>(p1, 128, NROWS, bnmean, bnvar);
  bn_apply_ker<_Float16><<<8192, 256, 0, stream>>>(p1, 128, (size_t)NROWS * 128,
                                                   bnmean, bnvar, e_g[0], e_bb[0], 1);
  _Float16* p2 = bigC;  // 65536 x 256
  gemm_wrow<_Float16, _Float16><<<gRows, GEMM_T, 0, stream>>>(
      p1, 128, e_w16[1], 128, e_b[1], nullptr, 0, p2, 256, NROWS, 256, 128, 0, 1.f);
  bn_stats_ker<_Float16><<<256, 256, 0, stream>>>(p2, 256, NROWS, bnmean, bnvar);
  bn_apply_ker<_Float16><<<8192, 256, 0, stream>>>(p2, 256, (size_t)NROWS * 256,
                                                   bnmean, bnvar, e_g[1], e_bb[1], 1);
  _Float16* p3 = bigB;  // 65536 x 512
  gemm_wrow<_Float16, _Float16><<<gRows, GEMM_T, 0, stream>>>(
      p2, 256, e_w16[2], 256, e_b[2], nullptr, 0, p3, 512, NROWS, 512, 256, 0, 1.f);
  bn_stats_ker<_Float16><<<512, 256, 0, stream>>>(p3, 512, NROWS, bnmean, bnvar);
  bn_apply_ker<_Float16><<<8192, 256, 0, stream>>>(p3, 512, (size_t)NROWS * 512,
                                                   bnmean, bnvar, e_g[2], e_bb[2], 1);
  gemm_wrow<_Float16, _Float16><<<dim3(NROWS / 16, 2), GEMM_T, 0, stream>>>(
      p3, 512, e_w16[3], 512, e_b[3], nullptr, 0, p4, NENC, NROWS, NENC, 512, 0, 1.f);
  bn_stats_ker<_Float16><<<NENC, 256, 0, stream>>>(p4, NENC, NROWS, bnmean, bnvar);
  bn_apply_ker<_Float16><<<8192, 256, 0, stream>>>(p4, NENC, (size_t)NROWS * NENC,
                                                   bnmean, bnvar, e_g[3], e_bb[3], 0);
  group_max_ker<_Float16><<<NBG, 256, 0, stream>>>(p4, NENC, NM, tokens, NENC);

  // ---- stage 4: token reduce + positional embedding ----
  gemm_wrow<float, float><<<gTok, GEMM_T, 0, stream>>>(
      tokens, NENC, reduce_w16, NENC, reduce_b, nullptr, 0, xbuf, NDIM, NBG, NDIM, NENC, 0, 1.f);
  gemm_wrow<float, _Float16><<<gTok, GEMM_T, 0, stream>>>(
      center, 3, pos1_w16, 3, pos1_b, nullptr, 0, poshid, 128, NBG, 128, 3, 2, 1.f);
  gemm_wrow<_Float16, float><<<gTok, GEMM_T, 0, stream>>>(
      poshid, 128, pos2_w16, 128, pos2_b, nullptr, 0, posbuf, NDIM, NBG, NDIM, 128, 0, 1.f);

  // ---- stage 5: transformer ----
  const float scale = 0.125f;  // (384/6)^-0.5
  const int NTOKEL = NBG * NDIM;
  int slot = 0;
  for (int i = 0; i < NDEPTH; ++i) {
    add_ker<<<(NTOKEL + 255) / 256, 256, 0, stream>>>(xbuf, posbuf, hin, NTOKEL);
    layernorm_ker<<<NBG, 256, 0, stream>>>(hin, ln1_g + i * NDIM, ln1_b + i * NDIM,
                                           ybuf, NDIM);
    gemm_wrow<_Float16, _Float16><<<dim3(NBG / 16, 3), GEMM_T, 0, stream>>>(
        ybuf, NDIM, qkv_w16 + (size_t)i * QKVLD * NDIM, NDIM, nullptr, nullptr, 0,
        qkvbuf, QKVLD, NBG, QKVLD, NDIM, 0, 1.f);
    attn_scores_ker<<<dim3(NG / 16, 1, BB * NHEADS), 128, 0, stream>>>(qkvbuf, scores, scale);
    softmax_ker<<<BB * NHEADS * NG, 256, 0, stream>>>(scores, NG);
    attn_out_ker<<<dim3(NG / 16, 1, BB * NHEADS), 32, 0, stream>>>(scores, qkvbuf, attno);
    gemm_wrow<_Float16, float><<<gTok, GEMM_T, 0, stream>>>(
        attno, NDIM, proj_w16 + (size_t)i * NDIM * NDIM, NDIM, proj_b + i * NDIM,
        hin, NDIM, xbuf, NDIM, NBG, NDIM, NDIM, 0, 1.f);
    layernorm_ker<<<NBG, 256, 0, stream>>>(xbuf, ln2_g + i * NDIM, ln2_b + i * NDIM,
                                           ybuf, NDIM);
    gemm_wrow<_Float16, _Float16><<<dim3(NBG / 16, 3), GEMM_T, 0, stream>>>(
        ybuf, NDIM, fc1_w16 + (size_t)i * NHID * NDIM, NDIM, fc1_b + i * NHID,
        nullptr, 0, mlp, NHID, NBG, NHID, NDIM, 2, 1.f);
    gemm_wrow<_Float16, float><<<gTok, GEMM_T, 0, stream>>>(
        mlp, NHID, fc2_w16 + (size_t)i * NDIM * NHID, NHID, fc2_b + i * NDIM,
        xbuf, NDIM, xbuf, NDIM, NBG, NDIM, NHID, 0, 1.f);
    if (i == 3 || i == 7 || i == 11) {
      hipMemcpyAsync((float*)d_out + (size_t)slot * NTOKEL, xbuf,
                     (size_t)NTOKEL * sizeof(float), hipMemcpyDeviceToDevice, stream);
      ++slot;
    }
  }
}
